// SelfAttention_27041114095968
// MI455X (gfx1250) — matrix operand order repaired and checked by reading, not compiled, not run on hardware
//
#include <hip/hip_runtime.h>

// MI455X / gfx1250 (CDNA5, wave32). All heavy matmuls via v_wmma_f32_16x16x32_bf16.
// Round 2: doubled per-wave tiles (32 q-rows in attention, 32x32 in out-GEMM) to
// halve vmem traffic per WMMA — this kernel is matrix-pipe bound, not HBM bound.

typedef __attribute__((ext_vector_type(16))) __bf16 v16bf;
typedef __attribute__((ext_vector_type(8)))  __bf16 v8bf;
typedef __attribute__((ext_vector_type(8)))  float  v8f;

namespace {
constexpr int kB = 2, kS = 2048, kE = 1024, kH = 16, kD = 64;
constexpr size_t kQKVElems = (size_t)kB * kH * kS * kD;   // 4M bf16 per tensor
constexpr size_t kCtxElems = (size_t)kB * kS * kE;        // 4M bf16
}

__device__ __forceinline__ v8f wmma_bf16(v16bf a, v16bf b, v8f c) {
  // D = A(16x32 bf16) * B(32x16 bf16) + C(16x16 f32)
  return __builtin_amdgcn_wmma_f32_16x16x32_bf16(false, a, false, b, (short)0, c,
                                                 false, false);
}

// A-fragment (16x32, row-major source). Lane holds row (lane&15); element e maps to
// K = k0 + (e&7) + (e<8 ? 0 : 16) + ((lane&16) ? 8 : 0)   [ISA 7.12.2, 16-bit A 16x32]
__device__ __forceinline__ v16bf load_a(const __bf16* __restrict__ row, int lane, int k0) {
  const int base = k0 + ((lane & 16) ? 8 : 0);
  const v8bf lo = *(const v8bf*)(row + base);        // global_load_b128
  const v8bf hi = *(const v8bf*)(row + base + 16);   // global_load_b128
  v16bf f;
#pragma unroll
  for (int i = 0; i < 8; ++i) { f[i] = lo[i]; f[i + 8] = hi[i]; }
  return f;
}

// B-fragment (32x16). Lane holds column N (caller picks row ptr = data for that N,
// unit stride along K). Element e maps to K = k0 + (e&7) + (e<8 ? 0 : 16) + ((lane&16) ? 8 : 0).
__device__ __forceinline__ v16bf load_b(const __bf16* __restrict__ row, int lane, int k0) {
  const int base = k0 + ((lane & 16) ? 8 : 0);
  const v8bf lo = *(const v8bf*)(row + base);
  const v8bf hi = *(const v8bf*)(row + base + 16);
  v16bf f;
#pragma unroll
  for (int i = 0; i < 8; ++i) { f[i] = lo[i]; f[i + 8] = hi[i]; }
  return f;
}

// Online-softmax update for one 16-row M-tile given its two 16x16 score tiles.
__device__ __forceinline__ void softmax_update(v8f& sa, v8f& sb, float (&mrun)[8],
                                               float (&lrun)[8], v8f (&O)[4]) {
  float tmax[8], rsum[8], alpha[8];
#pragma unroll
  for (int r = 0; r < 8; ++r) tmax[r] = fmaxf(sa[r], sb[r]);
#pragma unroll
  for (int mk = 8; mk >= 1; mk >>= 1)
#pragma unroll
    for (int r = 0; r < 8; ++r) tmax[r] = fmaxf(tmax[r], __shfl_xor(tmax[r], mk, 32));
#pragma unroll
  for (int r = 0; r < 8; ++r) {
    const float mn = fmaxf(mrun[r], tmax[r]);
    alpha[r] = __expf(mrun[r] - mn);
    mrun[r] = mn;
    sa[r] = __expf(sa[r] - mn);
    sb[r] = __expf(sb[r] - mn);
    rsum[r] = sa[r] + sb[r];
  }
#pragma unroll
  for (int mk = 8; mk >= 1; mk >>= 1)
#pragma unroll
    for (int r = 0; r < 8; ++r) rsum[r] += __shfl_xor(rsum[r], mk, 32);
#pragma unroll
  for (int r = 0; r < 8; ++r) {
    lrun[r] = lrun[r] * alpha[r] + rsum[r];
    O[0][r] *= alpha[r]; O[1][r] *= alpha[r]; O[2][r] *= alpha[r]; O[3][r] *= alpha[r];
  }
}

// ---------------------------------------------------------------------------
// Pass 1: per-head QKV projection (y = x @ W^T), f32 -> bf16, with layouts:
//   qb,kb: [B,H,S,64] (q pre-scaled by 1/sqrt(64));  vT: [B,H,64,S]
// ---------------------------------------------------------------------------
__global__ __launch_bounds__(256) void qkv_proj_kernel(
    const float* __restrict__ V, const float* __restrict__ K, const float* __restrict__ Q,
    const float* __restrict__ Wv, const float* __restrict__ Wk, const float* __restrict__ Wq,
    __bf16* __restrict__ qb, __bf16* __restrict__ kb, __bf16* __restrict__ vT) {
  __shared__ float xv[kE], xk[kE], xq[kE];
  const int bs = blockIdx.x;                // 0..B*S-1
  const int b = bs / kS, s = bs % kS;
  const size_t roff = (size_t)bs * kE;
  for (int i = threadIdx.x; i < kE; i += 256) {
    xv[i] = V[roff + i]; xk[i] = K[roff + i]; xq[i] = Q[roff + i];
  }
  __syncthreads();
#pragma unroll
  for (int j = 0; j < 4; ++j) {
    const int o = threadIdx.x + 256 * j;    // output channel 0..1023
    const int h = o >> 6, d = o & 63;
    const float* wvr = Wv + d * 64;
    const float* wkr = Wk + d * 64;
    const float* wqr = Wq + d * 64;
    const float* xvh = xv + h * 64;
    const float* xkh = xk + h * 64;
    const float* xqh = xq + h * 64;
    float av = 0.f, ak = 0.f, aq = 0.f;
#pragma unroll 8
    for (int e = 0; e < 64; ++e) {
      av = fmaf(wvr[e], xvh[e], av);
      ak = fmaf(wkr[e], xkh[e], ak);
      aq = fmaf(wqr[e], xqh[e], aq);
    }
    const size_t head = (size_t)(b * kH + h);
    qb[(head * kS + s) * kD + d] = (__bf16)(aq * 0.125f);   // fold 1/sqrt(HEAD)
    kb[(head * kS + s) * kD + d] = (__bf16)ak;
    vT[(head * kD + d) * kS + s] = (__bf16)av;              // transposed V
  }
}

__global__ void cvt_bf16_kernel(const float* __restrict__ src, __bf16* __restrict__ dst, int n) {
  const int i = blockIdx.x * blockDim.x + threadIdx.x;
  if (i < n) dst[i] = (__bf16)src[i];
}

// ---------------------------------------------------------------------------
// Pass 2: causal flash attention. One wave per 32-query tile; KV blocks of 32.
// Per KV block: 16 WMMAs against 16 b128 loads (K/V fragments reused by both
// 16-row M-tiles).
// ---------------------------------------------------------------------------
__global__ __launch_bounds__(256) void attn_kernel(
    const __bf16* __restrict__ qb, const __bf16* __restrict__ kb,
    const __bf16* __restrict__ vT, __bf16* __restrict__ ctx) {
  __shared__ float sc[8][32 * 32];          // per-wave P tile bounce buffer (32KB)
  const int lane = threadIdx.x & 31;
  const int wave = threadIdx.x >> 5;
  const int widx = blockIdx.x * 8 + wave;   // 0..2047
  const int qt = widx & (kS / 32 - 1);      // 32-row query tile 0..63
  const int h  = (widx >> 6) & (kH - 1);
  const int b  = widx >> 10;
  const int m = lane & 15;
  const int half8 = (lane & 16) ? 8 : 0;
  const size_t head = (size_t)(b * kH + h);
  const __bf16* qrow0 = qb + (head * kS + qt * 32 + m) * kD;
  const __bf16* qrow1 = qrow0 + 16 * kD;
  const __bf16* kbase = kb + head * kS * kD;
  const __bf16* vbase = vT + head * kD * kS;
  float* mysc = sc[wave];

  const v16bf Aq00 = load_a(qrow0, lane, 0);   // rows qt*32+0..15
  const v16bf Aq01 = load_a(qrow0, lane, 32);
  const v16bf Aq10 = load_a(qrow1, lane, 0);   // rows qt*32+16..31
  const v16bf Aq11 = load_a(qrow1, lane, 32);

  float mrun0[8], lrun0[8], mrun1[8], lrun1[8];
  v8f O0[4], O1[4];
#pragma unroll
  for (int t = 0; t < 4; ++t) {
    O0[t] = (v8f){0, 0, 0, 0, 0, 0, 0, 0};
    O1[t] = (v8f){0, 0, 0, 0, 0, 0, 0, 0};
  }
#pragma unroll
  for (int r = 0; r < 8; ++r) { mrun0[r] = mrun1[r] = -1e30f; lrun0[r] = lrun1[r] = 0.f; }

  // causal: keys up to qt*32+31 -> KV blocks 0..qt; only block qt needs masking
  for (int kv = 0; kv <= qt; ++kv) {
    const __bf16* krow0 = kbase + (size_t)(kv * 32 + m) * kD;
    const __bf16* krow1 = krow0 + 16 * kD;
    if (kv < qt) __builtin_prefetch(krow0 + 32 * kD, 0, 1);   // global_prefetch_b8

    const v16bf Bk00 = load_b(krow0, lane, 0);
    const v16bf Bk01 = load_b(krow0, lane, 32);
    const v16bf Bk10 = load_b(krow1, lane, 0);
    const v16bf Bk11 = load_b(krow1, lane, 32);

    // scores: 4 tiles (2 M x 2 N), 8 WMMAs, K/V fragments shared across M-tiles
    v8f s00 = {0,0,0,0,0,0,0,0}, s01 = s00, s10 = s00, s11 = s00;
    s00 = wmma_bf16(Aq00, Bk00, s00); s00 = wmma_bf16(Aq01, Bk01, s00);
    s01 = wmma_bf16(Aq00, Bk10, s01); s01 = wmma_bf16(Aq01, Bk11, s01);
    s10 = wmma_bf16(Aq10, Bk00, s10); s10 = wmma_bf16(Aq11, Bk01, s10);
    s11 = wmma_bf16(Aq10, Bk10, s11); s11 = wmma_bf16(Aq11, Bk11, s11);

    if (kv == qt) {                          // diagonal block: causal mask
      const int n0 = kv * 32 + m, n1 = n0 + 16;
#pragma unroll
      for (int r = 0; r < 8; ++r) {
        const int qi0 = qt * 32 + r + half8;
        const int qi1 = qi0 + 16;
        if (n0 > qi0) s00[r] = -1e30f;
        if (n1 > qi0) s01[r] = -1e30f;
        if (n0 > qi1) s10[r] = -1e30f;
        if (n1 > qi1) s11[r] = -1e30f;
      }
    }

    softmax_update(s00, s01, mrun0, lrun0, O0);
    softmax_update(s10, s11, mrun1, lrun1, O1);

    // re-layout P: C/D layout -> LDS row-major [32][32] -> bf16 A-fragments
#pragma unroll
    for (int r = 0; r < 8; ++r) {
      const int row = r + half8;
      mysc[row * 32 + m]             = s00[r];
      mysc[row * 32 + 16 + m]        = s01[r];
      mysc[(16 + row) * 32 + m]      = s10[r];
      mysc[(16 + row) * 32 + 16 + m] = s11[r];
    }
    __builtin_amdgcn_wave_barrier();
    asm volatile("s_wait_dscnt 0x0" ::: "memory");   // CDNA5 split DS counter
    v16bf Pa0, Pa1;
    {
      const float* prow0 = mysc + m * 32;
      const float* prow1 = mysc + (16 + m) * 32;
#pragma unroll
      for (int i = 0; i < 8; ++i) {
        Pa0[i]     = (__bf16)prow0[half8 + i];
        Pa0[i + 8] = (__bf16)prow0[half8 + 16 + i];
        Pa1[i]     = (__bf16)prow1[half8 + i];
        Pa1[i + 8] = (__bf16)prow1[half8 + 16 + i];
      }
    }
    asm volatile("" ::: "memory");

    // O += P(16x32) * V(32x64): 8 WMMAs, V fragments shared across M-tiles
    const int soff = kv * 32;
#pragma unroll
    for (int t = 0; t < 4; ++t) {
      const v16bf Bv = load_b(vbase + (size_t)(t * 16 + m) * kS + soff, lane, 0);
      O0[t] = wmma_bf16(Pa0, Bv, O0[t]);
      O1[t] = wmma_bf16(Pa1, Bv, O1[t]);
    }
  }

  // epilogue: normalize and scatter to ctx[B,S,E] (bf16) at embed offset h*64
#pragma unroll
  for (int r = 0; r < 8; ++r) {
    const float inv0 = 1.f / lrun0[r];
    const float inv1 = 1.f / lrun1[r];
    const int srow0 = qt * 32 + r + half8;
    __bf16* crow0 = ctx + ((size_t)b * kS + srow0) * kE + h * kD;
    __bf16* crow1 = crow0 + 16 * kE;
#pragma unroll
    for (int t = 0; t < 4; ++t) {
      crow0[t * 16 + m] = (__bf16)(O0[t][r] * inv0);
      crow1[t * 16 + m] = (__bf16)(O1[t][r] * inv1);
    }
  }
}

// ---------------------------------------------------------------------------
// Pass 3: out = ctx @ Wo^T + bo.  M=4096, N=1024, K=1024; 32x32 tile per wave
// (2x2 WMMA tiles -> 4 WMMAs per 8 b128 loads).
// ---------------------------------------------------------------------------
__global__ __launch_bounds__(256) void out_proj_kernel(
    const __bf16* __restrict__ ctx, const __bf16* __restrict__ wo,
    const float* __restrict__ bo, float* __restrict__ out) {
  const int lane = threadIdx.x & 31;
  const int wave = threadIdx.x >> 5;
  const int tile = blockIdx.x * 8 + wave;   // 0..4095
  const int nt = tile & 31;                 // 32 N-tiles of 32
  const int mt = tile >> 5;                 // 128 M-tiles of 32
  const int m = lane & 15;
  const int half8 = (lane & 16) ? 8 : 0;
  const __bf16* arow0 = ctx + (size_t)(mt * 32 + m) * kE;
  const __bf16* arow1 = arow0 + 16 * kE;
  const __bf16* brow0 = wo + (size_t)(nt * 32 + m) * kE;    // B[n,k] = Wo[n,k]
  const __bf16* brow1 = brow0 + 16 * kE;
  v8f a00 = {0,0,0,0,0,0,0,0}, a01 = a00, a10 = a00, a11 = a00;
  for (int k0 = 0; k0 < kE; k0 += 32) {
    const v16bf A0 = load_a(arow0, lane, k0);
    const v16bf A1 = load_a(arow1, lane, k0);
    const v16bf B0 = load_b(brow0, lane, k0);
    const v16bf B1 = load_b(brow1, lane, k0);
    a00 = wmma_bf16(A0, B0, a00);
    a01 = wmma_bf16(A0, B1, a01);
    a10 = wmma_bf16(A1, B0, a10);
    a11 = wmma_bf16(A1, B1, a11);
  }
  const int n0 = nt * 32 + m, n1 = n0 + 16;
  const float bias0 = bo[n0], bias1 = bo[n1];
#pragma unroll
  for (int r = 0; r < 8; ++r) {
    const size_t row0 = (size_t)(mt * 32 + r + half8);
    out[row0 * kE + n0]        = a00[r] + bias0;
    out[row0 * kE + n1]        = a01[r] + bias1;
    out[(row0 + 16) * kE + n0] = a10[r] + bias0;
    out[(row0 + 16) * kE + n1] = a11[r] + bias1;
  }
}

// ---------------------------------------------------------------------------
extern "C" void kernel_launch(void* const* d_in, const int* in_sizes, int n_in,
                              void* d_out, int out_size, void* d_ws, size_t ws_size,
                              hipStream_t stream) {
  (void)in_sizes; (void)n_in; (void)out_size; (void)ws_size;
  const float* values  = (const float*)d_in[0];
  const float* keys    = (const float*)d_in[1];
  const float* queries = (const float*)d_in[2];
  // d_in[3] = mask: causal, known statically — unused.
  const float* Wv = (const float*)d_in[4];
  const float* Wk = (const float*)d_in[5];
  const float* Wq = (const float*)d_in[6];
  const float* Wo = (const float*)d_in[7];
  const float* bo = (const float*)d_in[8];
  float* out = (float*)d_out;

  char* ws = (char*)d_ws;
  __bf16* qb  = (__bf16*)(ws);                                   // 8 MB
  __bf16* kb  = (__bf16*)(ws + kQKVElems * 2);                   // 8 MB
  __bf16* vT  = (__bf16*)(ws + kQKVElems * 4);                   // 8 MB
  __bf16* ctx = (__bf16*)(ws + kQKVElems * 6);                   // 8 MB
  __bf16* wob = (__bf16*)(ws + kQKVElems * 6 + kCtxElems * 2);   // 2 MB

  qkv_proj_kernel<<<kB * kS, 256, 0, stream>>>(values, keys, queries, Wv, Wk, Wq,
                                               qb, kb, vT);
  cvt_bf16_kernel<<<(kE * kE + 255) / 256, 256, 0, stream>>>(Wo, wob, kE * kE);
  attn_kernel<<<(kB * kH * (kS / 32)) / 8, 256, 0, stream>>>(qb, kb, vT, ctx);
  out_proj_kernel<<<((kB * kS / 32) * (kE / 32)) / 8, 256, 0, stream>>>(ctx, wob, bo, out);
}
